// LinearAttention_heads_10857677324403
// MI455X (gfx1250) — matrix operand order repaired and checked by reading, not compiled, not run on hardware
//
#include <hip/hip_runtime.h>
#include <hip/hip_bf16.h>
#include <stdint.h>

// ---------------------------------------------------------------------------
// Linformer attention for MI455X (gfx1250, wave32, WMMA bf16 + TDM path)
//   B=32, N=4096, D=128, Kp=256
// ---------------------------------------------------------------------------

#define B_SZ  32
#define N_SZ  4096
#define D_SZ  128
#define KP_SZ 256

typedef __bf16 v16bf __attribute__((ext_vector_type(16)));
typedef float  v8f   __attribute__((ext_vector_type(8)));
typedef unsigned int v4u __attribute__((ext_vector_type(4)));
typedef int    v8i   __attribute__((ext_vector_type(8)));
typedef int    v4i   __attribute__((ext_vector_type(4)));

#if defined(__has_builtin)
#  if __has_builtin(__builtin_amdgcn_tensor_load_to_lds)
#    define HAVE_TDM 1
#  endif
#endif
#ifndef HAVE_TDM
#  define HAVE_TDM 0
#endif
#if __has_include(<hip/amd_detail/amd_gfx1250_TDM.h>)
#  define TDM_6ARG 1   // amdgpu-toolchain (therock headers): 6-arg builtin
#else
#  define TDM_6ARG 0   // ROCm 7.2: 5-arg builtin
#endif

static __device__ __forceinline__ v8f wmma_bf16(v16bf a, v16bf b, v8f c) {
    // D = A(16x32 bf16) * B(32x16 bf16) + C(16x16 f32)
    return __builtin_amdgcn_wmma_f32_16x16x32_bf16(
        /*neg_a=*/false, a, /*neg_b=*/false, b,
        /*c_mod=*/(short)0, c, /*reuse_a=*/false, /*reuse_b=*/false);
}

// B-operand: 16 contiguous bf16 (fixed column, K-range (lane/16)*16 .. +15)
static __device__ __forceinline__ v16bf load16bf(const __bf16* p) {
    union U { uint4 u; __bf16 h[8]; };
    U a, b;
    a.u = ((const uint4*)p)[0];
    b.u = ((const uint4*)p)[1];
    v16bf r;
#pragma unroll
    for (int i = 0; i < 8; ++i) { r[i] = a.h[i]; r[i + 8] = b.h[i]; }
    return r;
}

// A-operand from fp32 row: contiguous p[0..15]  (p = row + kbase + half*16)
static __device__ __forceinline__ v16bf makeA_f32(const float* p) {
    float4 c0 = *(const float4*)(p);
    float4 c1 = *(const float4*)(p + 4);
    float4 c2 = *(const float4*)(p + 8);
    float4 c3 = *(const float4*)(p + 12);
    v16bf r;
    r[0] = (__bf16)c0.x; r[1] = (__bf16)c0.y; r[2]  = (__bf16)c0.z; r[3]  = (__bf16)c0.w;
    r[4] = (__bf16)c1.x; r[5] = (__bf16)c1.y; r[6]  = (__bf16)c1.z; r[7]  = (__bf16)c1.w;
    r[8] = (__bf16)c2.x; r[9] = (__bf16)c2.y; r[10] = (__bf16)c2.z; r[11] = (__bf16)c2.w;
    r[12]= (__bf16)c3.x; r[13]= (__bf16)c3.y; r[14] = (__bf16)c3.z; r[15] = (__bf16)c3.w;
    return r;
}

// A-operand from bf16 (LDS): contiguous p[0..15]
static __device__ __forceinline__ v16bf makeA_bf(const __bf16* p) {
    union U { uint4 u; __bf16 h[8]; };
    U a, b;
    a.u = *(const uint4*)(p);
    b.u = *(const uint4*)(p + 8);
    v16bf r;
#pragma unroll
    for (int i = 0; i < 8; ++i) { r[i] = a.h[i]; r[i + 8] = b.h[i]; }
    return r;
}

#if HAVE_TDM
// Issue one TDM 2D tile load: global (rows x cols, bf16, row-major, contiguous)
// -> LDS.  D# packed per CDNA5 ISA.
static __device__ __forceinline__ void tdm_load_2d_bf16(
    const void* gptr, void* lptr, uint32_t rows, uint32_t cols)
{
    const uint64_t ga = (uint64_t)(uintptr_t)gptr;
    const uint32_t la = (uint32_t)(uintptr_t)lptr;  // flat LDS addr truncates to offset

    v4u g0;
    g0[0] = 1u;                                          // count=1, user D#
    g0[1] = la;                                          // lds_addr
    g0[2] = (uint32_t)(ga & 0xFFFFFFFFu);                // global_addr[31:0]
    g0[3] = (uint32_t)((ga >> 32) & 0x01FFFFFFu) | (2u << 30);  // ga[56:32] | type=2

    v8i g1;
    g1[0] = (int)(1u << 16);                 // data_size = 2 bytes; no multicast
    g1[1] = (int)(cols << 16);               // tensor_dim0[15:0]
    g1[2] = (int)((cols >> 16) | (rows << 16)); // tensor_dim0[31:16] | tensor_dim1[15:0]
    g1[3] = (int)((rows >> 16) | (cols << 16)); // tensor_dim1[31:16] | tile_dim0
    g1[4] = (int)rows;                       // tile_dim1 (tile_dim2 = 0)
    g1[5] = (int)cols;                       // tensor_dim0_stride[31:0]
    g1[6] = 0;
    g1[7] = 0;

    v4i gz = {0, 0, 0, 0};
#if TDM_6ARG
    v8i gz8 = {0, 0, 0, 0, 0, 0, 0, 0};
    __builtin_amdgcn_tensor_load_to_lds(g0, g1, gz, gz, gz8, 0);
#else
    __builtin_amdgcn_tensor_load_to_lds(g0, g1, gz, gz, 0);
#endif
}
#endif

// ---------------------------------------------------------------------------
// Kernel 1: O = W @ X[b] + bias   (W: [256,4096] fp32, X: [B,4096,128] fp32)
//   transposeOut==0 : Out[b][k][d]  bf16 (K_projT, B-operand layout for QK gemm)
//   transposeOut==1 : Out[b][d][k]  bf16 (V_projT, B-operand layout for PV gemm)
// ---------------------------------------------------------------------------
__global__ __launch_bounds__(128) void proj_gemm(
    const float* __restrict__ W, const float* __restrict__ bias,
    const float* __restrict__ X, __bf16* __restrict__ Out, int transposeOut)
{
    __shared__ __bf16 xt[D_SZ][32];              // X chunk, transposed, bf16 (8 KB)

    const int b    = blockIdx.x >> 2;
    const int kq   = blockIdx.x & 3;
    const int wave = threadIdx.x >> 5;
    const int lane = threadIdx.x & 31;
    const int half = lane >> 4;
    const int l16  = lane & 15;

    const int   kRowTile = kq * 64 + wave * 16;
    const float* Wrow    = W + (size_t)(kRowTile + l16) * N_SZ;
    const float* Xb      = X + (size_t)b * N_SZ * D_SZ;

    v8f acc[8];
#pragma unroll
    for (int dt = 0; dt < 8; ++dt)
#pragma unroll
        for (int r = 0; r < 8; ++r) acc[dt][r] = 0.0f;

    for (int n0 = 0; n0 < N_SZ; n0 += 32) {
        __syncthreads();
        {   // cooperative stage: X[n0..n0+31][0..127] -> xt[d][n] bf16
            const int dq = (threadIdx.x & 31) * 4;
            const int nr = threadIdx.x >> 5;
#pragma unroll
            for (int i = 0; i < 8; ++i) {
                const int n = nr + i * 4;
                float4 v = *(const float4*)(Xb + (size_t)(n0 + n) * D_SZ + dq);
                xt[dq + 0][n] = (__bf16)v.x;
                xt[dq + 1][n] = (__bf16)v.y;
                xt[dq + 2][n] = (__bf16)v.z;
                xt[dq + 3][n] = (__bf16)v.w;
            }
        }
        __syncthreads();

        v16bf a = makeA_f32(Wrow + n0 + half * 16);
#pragma unroll
        for (int dt = 0; dt < 8; ++dt) {
            v16bf bm = load16bf(&xt[dt * 16 + l16][half * 16]);
            acc[dt] = wmma_bf16(a, bm, acc[dt]);
        }
    }

    const int mbase = kRowTile + half * 8;
    float bv[8];
#pragma unroll
    for (int r = 0; r < 8; ++r) bv[r] = bias[mbase + r];

    if (!transposeOut) {
        __bf16* Ob = Out + (size_t)b * KP_SZ * D_SZ;
#pragma unroll
        for (int dt = 0; dt < 8; ++dt)
#pragma unroll
            for (int r = 0; r < 8; ++r)
                Ob[(size_t)(mbase + r) * D_SZ + dt * 16 + l16] =
                    (__bf16)(acc[dt][r] + bv[r]);
    } else {
        __bf16* Ob = Out + (size_t)b * D_SZ * KP_SZ;
#pragma unroll
        for (int dt = 0; dt < 8; ++dt)
#pragma unroll
            for (int r = 0; r < 8; ++r)
                Ob[(size_t)(dt * 16 + l16) * KP_SZ + mbase + r] =
                    (__bf16)(acc[dt][r] + bv[r]);
    }
}

// ---------------------------------------------------------------------------
// Kernel 2: per 16-row tile of Q:
//   S = Q_tile @ K_projT^T * scale ; P = softmax_k(S) ; O = P @ V_proj
// Block = 128 thr (4 waves, same batch). K_projT / V_projT staged into LDS once
// per block via TDM (V_projT DMA overlaps the QK GEMM + softmax). P staging is
// aliased onto the K_projT tile (dead after GEMM1, barrier-enforced).
// ---------------------------------------------------------------------------
__global__ __launch_bounds__(128) void linformer_attn(
    const float*  __restrict__ Q,     // [B,4096,128] fp32
    const __bf16* __restrict__ KpT,   // [B,256,128]  bf16
    const __bf16* __restrict__ VpT,   // [B,128,256]  bf16
    float*        __restrict__ Out)   // [B,4096,128] fp32
{
    __shared__ union {
        __bf16 kp[KP_SZ][D_SZ];          // 64 KB : K_projT tile (phase 1)
        __bf16 p[4][16][KP_SZ];          // 32 KB : P rows        (phase 2)
    } sA;
    __shared__ __bf16 sVp[D_SZ][KP_SZ];  // 64 KB : V_projT tile

    const int wave = threadIdx.x >> 5;
    const int lane = threadIdx.x & 31;
    const int half = lane >> 4;
    const int l16  = lane & 15;

    const int tile    = blockIdx.x * 4 + wave;   // 4 tiles/block, same batch
    const int b       = tile >> 8;
    const int rowBase = (tile & 255) * 16;

    const float*  Qrow = Q   + ((size_t)b * N_SZ + rowBase + l16) * D_SZ;
    const __bf16* Kpb  = KpT + (size_t)b * KP_SZ * D_SZ;
    const __bf16* Vpb  = VpT + (size_t)b * D_SZ * KP_SZ;

#if HAVE_TDM
    if (wave == 0) {
        tdm_load_2d_bf16(Kpb, &sA.kp[0][0], KP_SZ, D_SZ);   // needed first
        tdm_load_2d_bf16(Vpb, &sVp[0][0],   D_SZ, KP_SZ);   // overlaps GEMM1
    }
#endif

    // --- Q row-block as 4 A-tiles (overlaps the TDM transfers) ---
    v16bf aq[4];
#pragma unroll
    for (int t = 0; t < 4; ++t) aq[t] = makeA_f32(Qrow + t * 32 + half * 16);

    v8f sc[16];
#pragma unroll
    for (int ct = 0; ct < 16; ++ct)
#pragma unroll
        for (int r = 0; r < 8; ++r) sc[ct][r] = 0.0f;

#if HAVE_TDM
    if (wave == 0) {
        __builtin_amdgcn_s_wait_tensorcnt(1);   // K_projT landed (V still in flight)
        asm volatile("" ::: "memory");
    }
#else
    {   // fallback: cooperative bf16 copies
        for (int i = threadIdx.x; i < KP_SZ * D_SZ / 8; i += 128)
            ((uint4*)&sA.kp[0][0])[i] = ((const uint4*)Kpb)[i];
        for (int i = threadIdx.x; i < D_SZ * KP_SZ / 8; i += 128)
            ((uint4*)&sVp[0][0])[i] = ((const uint4*)Vpb)[i];
    }
#endif
    __syncthreads();                             // K_projT visible to all waves

    // --- scores: 16 tiles of 16x16 (full 16x256 score block in VGPRs) ---
#pragma unroll
    for (int ct = 0; ct < 16; ++ct)
#pragma unroll
        for (int t = 0; t < 4; ++t) {
            v16bf bm = load16bf(&sA.kp[ct * 16 + l16][t * 32 + half * 16]);
            sc[ct] = wmma_bf16(aq[t], bm, sc[ct]);
        }

    __syncthreads();                             // K_projT dead -> reuse sA for P

    // --- softmax over k (256) per row; rows live in 16-lane groups ---
    const float scale = 0.088388347648318447f;   // 1/sqrt(128)
#pragma unroll
    for (int r = 0; r < 8; ++r) {
        float mx = -3.0e38f;
#pragma unroll
        for (int ct = 0; ct < 16; ++ct) mx = fmaxf(mx, sc[ct][r]);
#pragma unroll
        for (int m = 1; m < 16; m <<= 1) mx = fmaxf(mx, __shfl_xor(mx, m, 32));
        float s = 0.0f;
#pragma unroll
        for (int ct = 0; ct < 16; ++ct) {
            float e = __expf(scale * (sc[ct][r] - mx));
            sc[ct][r] = e;
            s += e;
        }
#pragma unroll
        for (int m = 1; m < 16; m <<= 1) s += __shfl_xor(s, m, 32);
        const float inv = 1.0f / s;
        const int   mrow = half * 8 + r;
#pragma unroll
        for (int ct = 0; ct < 16; ++ct)
            sA.p[wave][mrow][ct * 16 + l16] = (__bf16)(sc[ct][r] * inv);
    }

#if HAVE_TDM
    if (wave == 0) {
        __builtin_amdgcn_s_wait_tensorcnt(0);   // V_projT landed
        asm volatile("" ::: "memory");
    }
#endif
    __syncthreads();                             // P + V_projT visible

    // --- O = P @ V_proj : 8 output tiles, reduction over k=256 in steps of 32 ---
    v8f oacc[8];
#pragma unroll
    for (int dt = 0; dt < 8; ++dt)
#pragma unroll
        for (int r = 0; r < 8; ++r) oacc[dt][r] = 0.0f;

#pragma unroll
    for (int s = 0; s < 8; ++s) {
        v16bf ap = makeA_bf(&sA.p[wave][l16][s * 32 + half * 16]);
#pragma unroll
        for (int dt = 0; dt < 8; ++dt) {
            v16bf bm = load16bf(&sVp[dt * 16 + l16][s * 32 + half * 16]);
            oacc[dt] = wmma_bf16(ap, bm, oacc[dt]);
        }
    }

    // --- epilogue: fp32 store ---
    float* Orow = Out + ((size_t)b * N_SZ + rowBase) * D_SZ;
#pragma unroll
    for (int dt = 0; dt < 8; ++dt)
#pragma unroll
        for (int r = 0; r < 8; ++r)
            Orow[(size_t)(half * 8 + r) * D_SZ + dt * 16 + l16] = oacc[dt][r];
}

// ---------------------------------------------------------------------------
extern "C" void kernel_launch(void* const* d_in, const int* in_sizes, int n_in,
                              void* d_out, int out_size, void* d_ws, size_t ws_size,
                              hipStream_t stream) {
    (void)in_sizes; (void)n_in; (void)out_size; (void)ws_size;

    const float* Q   = (const float*)d_in[0];
    const float* K   = (const float*)d_in[1];
    const float* V   = (const float*)d_in[2];
    const float* E_W = (const float*)d_in[3];
    const float* E_b = (const float*)d_in[4];
    const float* F_W = (const float*)d_in[5];
    const float* F_b = (const float*)d_in[6];
    float* O = (float*)d_out;

    // workspace: K_projT [32,256,128] bf16 (2 MB) + V_projT [32,128,256] bf16 (2 MB)
    __bf16* KprojT = (__bf16*)d_ws;
    __bf16* VprojT = KprojT + (size_t)B_SZ * KP_SZ * D_SZ;

    proj_gemm<<<dim3(B_SZ * 4), dim3(128), 0, stream>>>(E_W, E_b, K, KprojT, 0);
    proj_gemm<<<dim3(B_SZ * 4), dim3(128), 0, stream>>>(F_W, F_b, V, VprojT, 1);
    linformer_attn<<<dim3(B_SZ * N_SZ / 16 / 4), dim3(128), 0, stream>>>(Q, KprojT, VprojT, O);
}
